// TCQE_39384850104407
// MI455X (gfx1250) — matrix operand order repaired and checked by reading, not compiled, not run on hardware
//
#include <hip/hip_runtime.h>
#include <hip/hip_bf16.h>

#define PI_F 3.14159265358979323846f
#define BQ   16384   // batch
#define DD   256     // entity dim
#define HH   800     // hidden dim
#define HHP  896     // hidden dim padded to multiple of 128

typedef __attribute__((ext_vector_type(16))) _Float16 v16h;
typedef __attribute__((ext_vector_type(8)))  _Float16 v8h;
typedef __attribute__((ext_vector_type(8)))  float    v8f;

__device__ __forceinline__ float to_axis_f(float x) { return tanhf(x) * PI_F; }
__device__ __forceinline__ float to_arg_f(float x)  { return tanhf(2.f * x) * (PI_F * 0.5f) + PI_F * 0.5f; }

// --- CDNA5 async global->LDS copy (ASYNCcnt path, bypasses VGPRs) ------------
__device__ __forceinline__ void async_ld_b128(unsigned lds_off, const _Float16* g) {
    asm volatile("global_load_async_to_lds_b128 %0, %1, off"
                 :: "v"(lds_off), "v"(g) : "memory");
}
__device__ __forceinline__ void wait_async4() {   // <=4 outstanding (prev buffer done)
    asm volatile("s_wait_asynccnt 0x4" ::: "memory");
}
__device__ __forceinline__ void wait_async0() {
    asm volatile("s_wait_asynccnt 0x0" ::: "memory");
}

// ---------------------------------------------------------------------------
// Weight prep: transpose + pad + f32->f16.  dst[n*Kp + k] = src[k*N + n] (0 pad)
// ---------------------------------------------------------------------------
__global__ void k_prep_weight(const float* __restrict__ src, _Float16* __restrict__ dst,
                              int K, int N, int Kp, int Np) {
    int i = blockIdx.x * 256 + threadIdx.x;
    if (i >= Kp * Np) return;
    int n = i / Kp, k = i - n * Kp;
    float v = (n < N && k < K) ? src[(size_t)k * N + n] : 0.f;
    dst[i] = (_Float16)v;
}

__global__ void k_prep_bias(const float* __restrict__ src, float* __restrict__ dst, int N, int Np) {
    int i = blockIdx.x * 256 + threadIdx.x;
    if (i < Np) dst[i] = (i < N) ? src[i] : 0.f;
}

// ---------------------------------------------------------------------------
// Gather + to_axis/to_arg, build projection input X = [qf+rf, ql+rl] as f16
// ---------------------------------------------------------------------------
__global__ void k_build_x(const int* __restrict__ eidx, const int* __restrict__ ridx,
                          const float* __restrict__ ent, const float* __restrict__ relf,
                          const float* __restrict__ rell, _Float16* __restrict__ X) {
    int b = blockIdx.x;
    int d = threadIdx.x;
    int e = eidx[b], r = ridx[b];
    float qf = to_axis_f(ent[(size_t)e * DD + d]);
    float rf = to_axis_f(relf[(size_t)r * DD + d]);
    float rl = to_arg_f(rell[(size_t)r * DD + d]);
    X[(size_t)b * (2 * DD) + d]      = (_Float16)(qf + rf);   // feature half (ql = 0)
    X[(size_t)b * (2 * DD) + DD + d] = (_Float16)rl;          // logic half
}

// ---------------------------------------------------------------------------
// WMMA GEMM: C[M,N] = act(A[M,K] @ Wt^T + bias);  Wt is [N][K] f16 (pre-transposed)
// M%128 == N%128 == 0, K % 64 == 0 (guaranteed by padding) -> zero predication.
// 256 threads = 8 waves; block tile 128x128; wave tile 32x64 = 2x4 WMMA accums.
// LDS double buffer filled by async global->LDS copies (ASYNCcnt); statically
// unrolled by 2 so buffer indices are compile-time constants (immediate DS
// offsets, no cndmask/branches in the steady state).
// ---------------------------------------------------------------------------
template <bool RELU, bool OUT32>
__global__ __launch_bounds__(256) void k_gemm(const _Float16* __restrict__ A,
                                              const _Float16* __restrict__ Wt,
                                              const float* __restrict__ bias,
                                              void* __restrict__ Cout,
                                              int M, int N, int K) {
    constexpr int BM = 128, BN = 128, BK = 32, PAD = 8;  // row stride 40 halves = 80B (16B-aligned)
    __shared__ _Float16 As[2][BM][BK + PAD];   // M x K tiles, row-major
    __shared__ _Float16 Bs[2][BN][BK + PAD];   // N x K tiles, row-major (B-frag major)

    const int t    = threadIdx.x;
    const int lane = t & 31;
    const int wave = t >> 5;
    const int wm   = wave >> 1;      // 0..3  (M subtiles of 32)
    const int wn   = wave & 1;       // 0..1  (N subtiles of 64)
    const int m0   = blockIdx.y * BM;
    const int n0   = blockIdx.x * BN;

    const int lm = lane & 15;
    const int hi = lane >> 4;        // lanes 0-15 vs 16-31

    // staging: both tiles are 128 rows x 32 halves; 2 threads/row, 32B each
    const int srow = t >> 1, skh = (t & 1) * 16;
    const _Float16* gA = A  + (size_t)(m0 + srow) * K + skh;
    const _Float16* gB = Wt + (size_t)(n0 + srow) * K + skh;
    // wave-relative LDS byte offsets of this thread's staging slots (loop-invariant)
    const unsigned aO0 = (unsigned)(size_t)&As[0][srow][skh];
    const unsigned aO1 = (unsigned)(size_t)&As[1][srow][skh];
    const unsigned bO0 = (unsigned)(size_t)&Bs[0][srow][skh];
    const unsigned bO1 = (unsigned)(size_t)&Bs[1][srow][skh];

    v8f acc[2][4] = {};

    auto issue = [&](int buf, int k0) {                  // 4 x b128 async copies
        unsigned ao = buf ? aO1 : aO0;
        unsigned bo = buf ? bO1 : bO0;
        async_ld_b128(ao,      gA + k0);
        async_ld_b128(ao + 16, gA + k0 + 8);
        async_ld_b128(bo,      gB + k0);
        async_ld_b128(bo + 16, gB + k0 + 8);
        __builtin_prefetch(gA + k0 + 2 * BK, 0, 1);      // speculative, 2 tiles ahead
        __builtin_prefetch(gB + k0 + 2 * BK, 0, 1);
    };
    auto compute = [&](int buf) {
        // A frags 16x32: lanes 0-15 -> K 0..7 & 16..23; lanes 16-31 -> K 8..15 & 24..31
        v16h afr[2];
#pragma unroll
        for (int sm = 0; sm < 2; ++sm) {
            int ar  = wm * 32 + sm * 16 + lm;
            int kb2 = hi * 8;
            ((v8h*)&afr[sm])[0] = *(const v8h*)&As[buf][ar][kb2];
            ((v8h*)&afr[sm])[1] = *(const v8h*)&As[buf][ar][kb2 + 16];
        }
        // B frags 32x16: lane = column; lanes 0-15 -> K 0..7 & 16..23; lanes 16-31 -> K 8..15 & 24..31
        v16h bfr[4];
#pragma unroll
        for (int sn = 0; sn < 4; ++sn) {
            int bc = wn * 64 + sn * 16 + lm;
            int kk = hi * 8;
            ((v8h*)&bfr[sn])[0] = *(const v8h*)&Bs[buf][bc][kk];
            ((v8h*)&bfr[sn])[1] = *(const v8h*)&Bs[buf][bc][kk + 16];
        }
#pragma unroll
        for (int sm = 0; sm < 2; ++sm)
#pragma unroll
            for (int sn = 0; sn < 4; ++sn)
                acc[sm][sn] = __builtin_amdgcn_wmma_f32_16x16x32_f16(
                    false, afr[sm], false, bfr[sn], (short)0, acc[sm][sn], false, false);
    };

    // K/BK is even for all layer shapes -> process tiles in pairs, last pair peeled.
    issue(0, 0);
    for (int k0 = 0; k0 < K - 2 * BK; k0 += 2 * BK) {
        issue(1, k0 + BK);      wait_async4();
        __syncthreads();  compute(0);  __syncthreads();
        issue(0, k0 + 2 * BK);  wait_async4();
        __syncthreads();  compute(1);  __syncthreads();
    }
    issue(1, K - BK);  wait_async4();
    __syncthreads();  compute(0);  __syncthreads();
    wait_async0();
    __syncthreads();  compute(1);

    // epilogue: bias + optional ReLU; C/D layout: VGPR v -> row v (lanes 0-15) / 8+v (16-31)
#pragma unroll
    for (int sm = 0; sm < 2; ++sm) {
#pragma unroll
        for (int sn = 0; sn < 4; ++sn) {
            int col = n0 + wn * 64 + sn * 16 + lm;
            float bv = bias[col];
#pragma unroll
            for (int v = 0; v < 8; ++v) {
                int row = m0 + wm * 32 + sm * 16 + hi * 8 + v;
                float c = acc[sm][sn][v] + bv;
                if (RELU) c = fmaxf(c, 0.f);
                if (OUT32) ((float*)Cout)[(size_t)row * N + col] = c;
                else       ((_Float16*)Cout)[(size_t)row * N + col] = (_Float16)c;
            }
        }
    }
}

// ---------------------------------------------------------------------------
// Projection post: f = to_axis(Y[:,:256]), l = to_arg(Y[:,256:]);
// ---------------------------------------------------------------------------
__global__ void k_post_proj(const _Float16* __restrict__ Y, float* __restrict__ axis,
                            float* __restrict__ arg, _Float16* __restrict__ L) {
    int b = blockIdx.x;
    int d = threadIdx.x;
    float f = to_axis_f((float)Y[(size_t)b * (2 * DD) + d]);
    float l = to_arg_f((float)Y[(size_t)b * (2 * DD) + DD + d]);
    axis[(size_t)b * DD + d] = f;
    arg[(size_t)b * DD + d]  = l;
    L[(size_t)b * (2 * DD) + d]      = (_Float16)(f - l);
    L[(size_t)b * (2 * DD) + DD + d] = (_Float16)(f + l);
}

__global__ void k_mean2_f16(const _Float16* __restrict__ a, const _Float16* __restrict__ b,
                            _Float16* __restrict__ o, int n) {
    int i = blockIdx.x * 256 + threadIdx.x;
    if (i < n) o[i] = (_Float16)(0.5f * ((float)a[i] + (float)b[i]));
}

// ---------------------------------------------------------------------------
// Final: softmax over 2 branches, polar combine, arctan fixup, gate, negation.
// ---------------------------------------------------------------------------
__global__ void k_final(const float* __restrict__ att0, const float* __restrict__ att1,
                        const float* __restrict__ ax0, const float* __restrict__ ax1,
                        const float* __restrict__ ag0, const float* __restrict__ ag1,
                        const float* __restrict__ gatel, float* __restrict__ out) {
    int i = blockIdx.x * 256 + threadIdx.x;   // over B*D
    float s0 = att0[i], s1 = att1[i];
    float m  = fmaxf(s0, s1);
    float e0 = __expf(s0 - m), e1 = __expf(s1 - m);
    float inv = 1.f / (e0 + e1);
    float w0 = e0 * inv, w1 = e1 * inv;
    float a0 = ax0[i], a1 = ax1[i];
    float x = w0 * __cosf(a0) + w1 * __cosf(a1);
    float y = w0 * __sinf(a0) + w1 * __sinf(a1);
    x = (fabsf(x) < 0.001f) ? 0.001f : x;
    float ax = atanf(y / x);
    if (x < 0.f) ax += (y >= 0.f) ? PI_F : -PI_F;
    float gate = 1.f / (1.f + __expf(-gatel[i]));
    float ar = fminf(ag0[i], ag1[i]) * gate;
    ax = (ax >= 0.f) ? ax - PI_F : ax + PI_F;   // negation
    ar = PI_F - ar;
    out[i]                   = ax;
    out[(size_t)BQ * DD + i] = ar;
}

// ---------------------------------------------------------------------------
// Host-side orchestration
// ---------------------------------------------------------------------------
static inline void* bump(char*& p, size_t bytes) {
    void* r = p;
    p += (bytes + 255) & ~(size_t)255;
    return r;
}

extern "C" void kernel_launch(void* const* d_in, const int* in_sizes, int n_in,
                              void* d_out, int out_size, void* d_ws, size_t ws_size,
                              hipStream_t stream) {
    (void)in_sizes; (void)n_in; (void)out_size; (void)ws_size;
    const int*   e_idx[2] = { (const int*)d_in[0], (const int*)d_in[2] };
    const int*   r_idx[2] = { (const int*)d_in[1], (const int*)d_in[3] };
    const float* ent  = (const float*)d_in[4];
    const float* relf = (const float*)d_in[5];
    const float* rell = (const float*)d_in[6];
    const float* W1 = (const float*)d_in[7],  *b1 = (const float*)d_in[8];
    const float* W2 = (const float*)d_in[9],  *b2 = (const float*)d_in[10];
    const float* W0 = (const float*)d_in[11], *b0 = (const float*)d_in[12];
    const float* Wax1 = (const float*)d_in[13], *bax1 = (const float*)d_in[14];
    const float* Wax2 = (const float*)d_in[15], *bax2 = (const float*)d_in[16];
    const float* Warg1 = (const float*)d_in[17], *barg1 = (const float*)d_in[18];
    const float* Warg2 = (const float*)d_in[19], *barg2 = (const float*)d_in[20];
    float* out = (float*)d_out;

    char* p = (char*)d_ws;
    // pre-transposed padded f16 weights: Wt[Np][Kp]
    _Float16* W1t   = (_Float16*)bump(p, (size_t)HHP * (2 * DD) * 2);  // [896][512]
    _Float16* W2t   = (_Float16*)bump(p, (size_t)HHP * HHP * 2);       // [896][896]
    _Float16* W0t   = (_Float16*)bump(p, (size_t)(2 * DD) * HHP * 2);  // [512][896]
    _Float16* Wax1t = (_Float16*)bump(p, (size_t)DD * (2 * DD) * 2);   // [256][512]
    _Float16* Wax2t = (_Float16*)bump(p, (size_t)DD * DD * 2);         // [256][256]
    _Float16* Warg1t= (_Float16*)bump(p, (size_t)DD * (2 * DD) * 2);
    _Float16* Warg2t= (_Float16*)bump(p, (size_t)DD * DD * 2);
    float*    b1p   = (float*)bump(p, (size_t)HHP * 4);
    float*    b2p   = (float*)bump(p, (size_t)HHP * 4);
    _Float16* X     = (_Float16*)bump(p, (size_t)BQ * 2 * DD * 2);
    _Float16* H1    = (_Float16*)bump(p, (size_t)BQ * HHP * 2);
    _Float16* H2    = (_Float16*)bump(p, (size_t)BQ * HHP * 2);
    _Float16* Y     = (_Float16*)bump(p, (size_t)BQ * 2 * DD * 2);
    _Float16* AT    = (_Float16*)bump(p, (size_t)BQ * DD * 2);
    float* axisF[2]; float* argF[2]; _Float16* L[2]; float* attL[2]; _Float16* argh[2];
    for (int n = 0; n < 2; ++n) {
        axisF[n] = (float*)bump(p, (size_t)BQ * DD * 4);
        argF[n]  = (float*)bump(p, (size_t)BQ * DD * 4);
        L[n]     = (_Float16*)bump(p, (size_t)BQ * 2 * DD * 2);
        attL[n]  = (float*)bump(p, (size_t)BQ * DD * 4);
        argh[n]  = (_Float16*)bump(p, (size_t)BQ * DD * 2);
    }
    _Float16* arghm = (_Float16*)bump(p, (size_t)BQ * DD * 2);
    float*    gateL = (float*)bump(p, (size_t)BQ * DD * 4);

    auto prep = [&](const float* src, _Float16* dst, int K, int N, int Kp, int Np) {
        int n = Kp * Np;
        k_prep_weight<<<(n + 255) / 256, 256, 0, stream>>>(src, dst, K, N, Kp, Np);
    };
    prep(W1, W1t, 2 * DD, HH, 2 * DD, HHP);
    prep(W2, W2t, HH, HH, HHP, HHP);
    prep(W0, W0t, HH, 2 * DD, HHP, 2 * DD);
    prep(Wax1, Wax1t, 2 * DD, DD, 2 * DD, DD);
    prep(Wax2, Wax2t, DD, DD, DD, DD);
    prep(Warg1, Warg1t, 2 * DD, DD, 2 * DD, DD);
    prep(Warg2, Warg2t, DD, DD, DD, DD);
    k_prep_bias<<<(HHP + 255) / 256, 256, 0, stream>>>(b1, b1p, HH, HHP);
    k_prep_bias<<<(HHP + 255) / 256, 256, 0, stream>>>(b2, b2p, HH, HHP);

    const dim3 blk(256);
    auto ggrid = [](int M, int Np) { return dim3(Np / 128, M / 128); };

    for (int n = 0; n < 2; ++n) {
        k_build_x<<<BQ, blk, 0, stream>>>(e_idx[n], r_idx[n], ent, relf, rell, X);
        // projection MLP (padded hidden width HHP; pad cols are exactly 0 after ReLU)
        k_gemm<true,  false><<<ggrid(BQ, HHP),    blk, 0, stream>>>(X,  W1t, b1p, H1, BQ, HHP,    2 * DD);
        k_gemm<true,  false><<<ggrid(BQ, HHP),    blk, 0, stream>>>(H1, W2t, b2p, H2, BQ, HHP,    HHP);
        k_gemm<false, false><<<ggrid(BQ, 2 * DD), blk, 0, stream>>>(H2, W0t, b0,  Y,  BQ, 2 * DD, HHP);
        k_post_proj<<<BQ, blk, 0, stream>>>(Y, axisF[n], argF[n], L[n]);
        // intersection head per branch
        k_gemm<true,  false><<<ggrid(BQ, DD), blk, 0, stream>>>(L[n], Wax1t,  bax1,  AT,      BQ, DD, 2 * DD);
        k_gemm<false, true ><<<ggrid(BQ, DD), blk, 0, stream>>>(AT,   Wax2t,  bax2,  attL[n], BQ, DD, DD);
        k_gemm<true,  false><<<ggrid(BQ, DD), blk, 0, stream>>>(L[n], Warg1t, barg1, argh[n], BQ, DD, 2 * DD);
    }
    k_mean2_f16<<<(BQ * DD + 255) / 256, blk, 0, stream>>>(argh[0], argh[1], arghm, BQ * DD);
    k_gemm<false, true><<<ggrid(BQ, DD), blk, 0, stream>>>(arghm, Warg2t, barg2, gateL, BQ, DD, DD);
    k_final<<<(BQ * DD + 255) / 256, blk, 0, stream>>>(attL[0], attL[1], axisF[0], axisF[1],
                                                       argF[0], argF[1], gateL, out);
}
